// DMCNN_Encoder_argProb_61134564491528
// MI455X (gfx1250) — matrix operand order repaired and checked by reading, compiled, not run on hardware
//
#include <hip/hip_runtime.h>
#include <hip/hip_bf16.h>
#include <math.h>

typedef _Float16 f16;
typedef __attribute__((ext_vector_type(16))) _Float16 v16h;
typedef __attribute__((ext_vector_type(8)))  _Float16 v8h;
typedef __attribute__((ext_vector_type(8)))  float    v8f;

#define SZ      4096
#define SENLEN  128
#define DWE     100
#define DPE     5
#define DEE     32
#define DRE     16
#define DC      128
#define CIN     270     // 100 + 5 + 5 + 32 + 8*16
#define CINP    288     // padded to 9 K-tiles of 32
#define KT      9       // CINP / 32
#define NKK     27      // 3 taps * 9 K-tiles
#define LROWS   130     // 128 positions + 1 zero pad row each side
#define OUTW    984     // 3*DC + 600

// ---------------------------------------------------------------------------
// Prep: repack conv_w (DC, CIN, 3) fp32 -> f16 [tap][DC][CINP] (zero padded)
// ---------------------------------------------------------------------------
__global__ void pack_weights_kernel(const float* __restrict__ conv_w,
                                    f16* __restrict__ wf16) {
    int idx = blockIdx.x * blockDim.x + threadIdx.x;
    const int total = 3 * DC * CINP;
    if (idx >= total) return;
    int k   = idx / (DC * CINP);
    int rem = idx % (DC * CINP);
    int o   = rem / CINP;
    int c   = rem % CINP;
    float v = (c < CIN) ? conv_w[(o * CIN + c) * 3 + k] : 0.0f;
    wf16[idx] = (f16)v;
}

static __device__ __forceinline__ v16h ld_frag_pair(const f16* lo, const f16* hi) {
    union { v16h v; v8h h[2]; } u;
    u.h[0] = *(const v8h*)lo;
    u.h[1] = *(const v8h*)hi;
    return u.v;
}

// ---------------------------------------------------------------------------
// Fused: gather/pack -> WMMA conv GEMM -> piecewise masked max pool -> tanh
// One workgroup (8 wave32) per sample.
// ---------------------------------------------------------------------------
__global__ __launch_bounds__(256)
void dmcnn_fused_kernel(const int* __restrict__ inp,
                        const int* __restrict__ pos1,
                        const int* __restrict__ pos2,
                        const int* __restrict__ loc,
                        const int* __restrict__ loc_mark,
                        const int* __restrict__ subtype,
                        const int* __restrict__ argRole,
                        const float* __restrict__ maskL,
                        const float* __restrict__ maskM,
                        const float* __restrict__ maskR,
                        const float* __restrict__ word_emb,
                        const float* __restrict__ pos_emb,
                        const float* __restrict__ event_emb,
                        const float* __restrict__ role_emb,
                        const f16*  __restrict__ wf16,    // [3][DC][CINP]
                        const float* __restrict__ conv_b,
                        float* __restrict__ out) {
    extern __shared__ f16 smem[];            // [LROWS][CINP] packed activations
    const int s   = blockIdx.x;
    const int tid = threadIdx.x;

    // ---- Phase 1: gather embeddings -> LDS (f16), zero pad rows/channels ----
    const int ev_idx = subtype[s];
    for (int idx = tid; idx < LROWS * CINP; idx += 256) {
        int row = idx / CINP;        // 0..129 ; row 0 and 129 are conv padding
        int c   = idx % CINP;
        float v = 0.0f;
        if (row >= 1 && row <= SENLEN && c < CIN) {
            int p = row - 1;
            if (c < DWE) {
                v = word_emb[inp[s * SENLEN + p] * DWE + c];
            } else if (c < DWE + DPE) {
                v = pos_emb[pos1[s * SENLEN + p] * DPE + (c - DWE)];
            } else if (c < DWE + 2 * DPE) {
                v = pos_emb[pos2[s * SENLEN + p] * DPE + (c - DWE - DPE)];
            } else if (c < DWE + 2 * DPE + DEE) {
                v = event_emb[ev_idx * DEE + (c - DWE - 2 * DPE)];
            } else {
                int rc = c - (DWE + 2 * DPE + DEE);   // 0..127
                v = role_emb[argRole[s * 8 + (rc >> 4)] * DRE + (rc & 15)];
            }
        }
        smem[idx] = (f16)v;
    }
    __syncthreads();

    // ---- Phase 2: conv as GEMM via v_wmma_f32_16x16x32_f16 ----
    const int wave    = tid >> 5;     // 0..7 -> DC tile (rows 16w..16w+15)
    const int lane    = tid & 31;
    const int lanemod = lane & 15;
    const int hsel    = lane >> 4;    // lane-half select

    v8f acc[8];
#pragma unroll
    for (int t = 0; t < 8; ++t) acc[t] = (v8f)(0.0f);

    // Per-lane bases.
    // A (ISA 7.12.2, 16x32 f16): VGPR0-3 : K = kb + hsel*8 .. +7 ; VGPR4-7 : +16
    const f16* Wlane = wf16 + (wave * 16 + lanemod) * CINP + hsel * 8;
    // B (32x16 f16): same K order as A: kb + hsel*8 .. +7 ; then +16.
    const f16* Xlane = smem + lanemod * CINP + hsel * 8;

#define LOAD_A(kk) \
    ld_frag_pair(Wlane + ((kk) / KT) * (DC * CINP) + ((kk) % KT) * 32, \
                 Wlane + ((kk) / KT) * (DC * CINP) + ((kk) % KT) * 32 + 16)

    v16h afr = LOAD_A(0);            // software-pipelined A fragment
#pragma unroll
    for (int kk = 0; kk < NKK; ++kk) {
        const int k  = kk / KT;
        const int kb = (kk % KT) * 32;
        // Prefetch next A fragment one full K-tile ahead of its use.
        v16h anext = afr;
        if (kk + 1 < NKK) anext = LOAD_A(kk + 1);
        // Stage all 8 B fragments in distinct registers.
        v16h bfr[8];
#pragma unroll
        for (int t = 0; t < 8; ++t) {
            const f16* Xr = Xlane + (t * 16 + k) * CINP + kb;
            bfr[t] = ld_frag_pair(Xr, Xr + 16);
        }
#pragma unroll
        for (int t = 0; t < 8; ++t) {
            acc[t] = __builtin_amdgcn_wmma_f32_16x16x32_f16(
                /*neg_a=*/false, afr, /*neg_b=*/false, bfr[t],
                /*c_mod=*/(short)0, acc[t],
                /*reuse_a=*/false, /*reuse_b=*/false);
        }
        afr = anext;
#if defined(__has_builtin)
#if __has_builtin(__builtin_amdgcn_sched_group_barrier)
        // Pin per-tile schedule: [2 VMEM reads (A for kk+1)] ->
        // [16 DS reads (B for kk)] -> [8 WMMA (kk)].
        // B fragments stay simultaneously live (distinct VGPRs, staircase
        // s_wait_dscnt), and every A load is in flight for a whole group
        // (16 DS issues + 8 WMMAs) before its first consuming WMMA.
        if (kk + 1 < NKK)
            __builtin_amdgcn_sched_group_barrier(0x0020, 2,  0); // VMEM read
        __builtin_amdgcn_sched_group_barrier(0x0100, 16, 0);     // DS read
        __builtin_amdgcn_sched_group_barrier(0x0008, 8,  0);     // MFMA/WMMA
#endif
#endif
    }
#undef LOAD_A

    // ---- Phase 3: piecewise masked max pool + tanh, fused in-register ----
    // C/D layout: VGPR i -> channel (wave*16 + i + hsel*8), lane -> position
    // (t*16 + lanemod). Reduce over positions: per-lane max over tiles, then
    // xor-shuffle reduce within the 16-lane half.
    const float* masks[3] = { maskL, maskM, maskR };
#pragma unroll
    for (int mi = 0; mi < 3; ++mi) {
        float mv[8];
#pragma unroll
        for (int t = 0; t < 8; ++t)
            mv[t] = masks[mi][s * SENLEN + t * 16 + lanemod];
#pragma unroll
        for (int i = 0; i < 8; ++i) {
            const int ch = wave * 16 + hsel * 8 + i;
            const float bias = conv_b[ch];
            float best = -3.0e38f;
#pragma unroll
            for (int t = 0; t < 8; ++t)
                best = fmaxf(best, (acc[t][i] + bias) * mv[t] + 1.0f);
            // reduce across the 16 lanes of this half (offsets stay in-half)
            best = fmaxf(best, __shfl_xor(best, 1, 32));
            best = fmaxf(best, __shfl_xor(best, 2, 32));
            best = fmaxf(best, __shfl_xor(best, 4, 32));
            best = fmaxf(best, __shfl_xor(best, 8, 32));
            if (lanemod == 0)
                out[(size_t)s * OUTW + mi * DC + ch] = tanhf(best - 1.0f);
        }
    }

    // ---- Phase 4: ragged loc pooling (600 outputs/sample) + tanh ----
    const int mark = loc_mark[s];
    for (int j = tid; j < 600; j += 256) {
        float v;
        if (j < 400) {
            int r = j / DWE, d = j % DWE;
            v = word_emb[loc[s * 16 + r] * DWE + d];
        } else if (j < 500) {
            int d = j - 400;
            float sum = 0.0f;
            for (int r = 4; r < 4 + mark; ++r)
                sum += word_emb[loc[s * 16 + r] * DWE + d];
            v = sum / (float)mark;
        } else {
            int d = j - 500;
            v = word_emb[loc[s * 16 + 4 + mark] * DWE + d];
        }
        out[(size_t)s * OUTW + 3 * DC + j] = tanhf(v);
    }
}

// ---------------------------------------------------------------------------
extern "C" void kernel_launch(void* const* d_in, const int* in_sizes, int n_in,
                              void* d_out, int out_size, void* d_ws, size_t ws_size,
                              hipStream_t stream) {
    const int*   inp       = (const int*)  d_in[0];
    const int*   pos1      = (const int*)  d_in[1];
    const int*   pos2      = (const int*)  d_in[2];
    const int*   loc       = (const int*)  d_in[3];
    const int*   loc_mark  = (const int*)  d_in[4];
    const int*   subtype   = (const int*)  d_in[5];
    const int*   argRole   = (const int*)  d_in[6];
    const float* maskL     = (const float*)d_in[7];
    const float* maskM     = (const float*)d_in[8];
    const float* maskR     = (const float*)d_in[9];
    const float* word_emb  = (const float*)d_in[10];
    const float* pos_emb   = (const float*)d_in[11];
    const float* event_emb = (const float*)d_in[12];
    const float* role_emb  = (const float*)d_in[13];
    const float* conv_w    = (const float*)d_in[14];
    const float* conv_b    = (const float*)d_in[15];
    float*       out       = (float*)d_out;

    // workspace: packed f16 weights [3][DC][CINP] = 221,184 bytes
    f16* wf16 = (f16*)d_ws;
    const int wtot = 3 * DC * CINP;
    pack_weights_kernel<<<(wtot + 255) / 256, 256, 0, stream>>>(conv_w, wf16);

    const size_t lds_bytes = (size_t)LROWS * CINP * sizeof(f16);  // 74,880 B
    dmcnn_fused_kernel<<<SZ, 256, lds_bytes, stream>>>(
        inp, pos1, pos2, loc, loc_mark, subtype, argRole,
        maskL, maskM, maskR, word_emb, pos_emb, event_emb, role_emb,
        wf16, conv_b, out);
}
